// NeuralRadianceField_549755814570
// MI455X (gfx1250) — matrix operand order repaired and checked by reading, not compiled, not run on hardware
//
#include <hip/hip_runtime.h>
#include <hip/hip_bf16.h>
#include <math.h>

typedef __attribute__((ext_vector_type(16))) _Float16 v16h;
typedef __attribute__((ext_vector_type(8)))  _Float16 v8h;
typedef __attribute__((ext_vector_type(8)))  float    v8f;

#define STR   264   // LDS row stride (halfs) for 256-wide activation buffers: 528B, 132 dw, 132%64=4 -> banks 4m distinct
#define STRP  104   // LDS row stride (halfs) for pe/de buffer: 208B, 52 dw, 52%64=52=4*13, 13 coprime 16 -> distinct

// ---------------------------------------------------------------------------
// WMMA helpers
// ---------------------------------------------------------------------------
__device__ __forceinline__ v8f wmma_f16(v16h a, v16h b, v8f c) {
  // D = A(16x32 f16) * B(32x16 f16) + C(16x16 f32)
  return __builtin_amdgcn_wmma_f32_16x16x32_f16(false, a, false, b, (short)0, c, false, false);
}

// A fragment (16-bit A 16x32 layout): lane group0 holds K=kb..kb+7 (elems 0-7)
// and K=kb+16..kb+23 (elems 8-15); lane group1 is +8.  p already includes
// row*stride + colbase + (lane>>4)*8.
__device__ __forceinline__ v16h loadA(const _Float16* p) {
  v8h lo = *(const v8h*)(p);
  v8h hi = *(const v8h*)(p + 16);
  return __builtin_shufflevector(lo, hi, 0,1,2,3,4,5,6,7,8,9,10,11,12,13,14,15);
}

// ---------------------------------------------------------------------------
// Generic layer engine: two M-tiles (32 points), nTiles x 16 outs, KT k-steps
// of 32.  A fragments are hoisted into registers ONCE per layer (KT*2 frags,
// <=160 VGPRs for KT=10) so the n-tile loop is pure B-stream + WMMA; this
// cuts LDS A-read traffic ~16x vs re-loading per n-tile.  Writes relu(f16)
// activations into an LDS buffer.
// ---------------------------------------------------------------------------
template<int KT, bool RELU>
__device__ __forceinline__ void gemmTile(
    const _Float16* __restrict__ W, const float* __restrict__ bias, int nTiles,
    const _Float16* const aP0[], const _Float16* const aP1[],
    _Float16* out, int lane)
{
  const int m16 = lane & 15, hi = lane >> 4;

  v16h af0[KT], af1[KT];
#pragma unroll
  for (int ks = 0; ks < KT; ++ks) {
    af0[ks] = loadA(aP0[ks]);
    af1[ks] = loadA(aP1[ks]);
  }

  for (int nt = 0; nt < nTiles; ++nt) {
    v8f acc0 = {}, acc1 = {};
    const _Float16* wp = W + (size_t)nt * KT * 512 + lane * 16;
    __builtin_prefetch(wp + (size_t)KT * 512, 0, 1);   // next n-tile's weights
#pragma unroll
    for (int ks = 0; ks < KT; ++ks) {
      v16h b = *(const v16h*)(wp + (size_t)ks * 512);
      acc0 = wmma_f16(af0[ks], b, acc0);
      acc1 = wmma_f16(af1[ks], b, acc1);
    }
    const float bv = bias[nt * 16 + m16];
    _Float16* o0 = out + (size_t)(hi * 8)      * STR + nt * 16 + m16;
    _Float16* o1 = out + (size_t)(16 + hi * 8) * STR + nt * 16 + m16;
#pragma unroll
    for (int r = 0; r < 8; ++r) {
      float v0 = acc0[r] + bv, v1 = acc1[r] + bv;
      if (RELU) { v0 = fmaxf(v0, 0.f); v1 = fmaxf(v1, 0.f); }
      o0[(size_t)r * STR] = (_Float16)v0;
      o1[(size_t)r * STR] = (_Float16)v1;
    }
  }
}

// ---------------------------------------------------------------------------
// Fused NeRF MLP: one wave32 per block, 32 points per wave.
// ---------------------------------------------------------------------------
__global__ void __launch_bounds__(32)
nerf_fused(const float* __restrict__ pts, const float* __restrict__ dirs,
           const _Float16* __restrict__ wts, const float* __restrict__ bias,
           float* __restrict__ out, int npts, int S)
{
  __shared__ __align__(16) _Float16 sA[32 * STR];
  __shared__ __align__(16) _Float16 sB[32 * STR];
  __shared__ __align__(16) _Float16 sP[32 * STRP];   // cols 0..63 pe(39+pad), 64..95 de(27+pad)

  const int lane = threadIdx.x;
  const int m16  = lane & 15, hi = lane >> 4;
  const int base = blockIdx.x * 32;

  // ---- per-lane embedding of one point ----
  int idx = base + lane; if (idx >= npts) idx = npts - 1;
  const float p3[3] = { pts[idx*3+0], pts[idx*3+1], pts[idx*3+2] };

  _Float16* pr = sP + (size_t)lane * STRP;
#pragma unroll
  for (int j = 0; j < STRP; ++j) pr[j] = (_Float16)0.f;   // zero padding cols
#pragma unroll
  for (int c = 0; c < 3; ++c) {
#pragma unroll
    for (int f = 0; f < 6; ++f) {
      float e = p3[c] * (float)(1 << f);
      pr[c*6 + f]      = (_Float16)sinf(e);
      pr[18 + c*6 + f] = (_Float16)cosf(e);
    }
    pr[36 + c] = (_Float16)p3[c];
  }
  {
    int bidx = idx / S;
    float dx = dirs[bidx*3+0], dy = dirs[bidx*3+1], dz = dirs[bidx*3+2];
    float inv = 1.f / fmaxf(sqrtf(dx*dx + dy*dy + dz*dz), 1e-12f);
    const float d3[3] = { dx*inv, dy*inv, dz*inv };
#pragma unroll
    for (int c = 0; c < 3; ++c) {
#pragma unroll
      for (int f = 0; f < 4; ++f) {
        float e = d3[c] * (float)(1 << f);
        pr[64 + c*4 + f]      = (_Float16)sinf(e);
        pr[64 + 12 + c*4 + f] = (_Float16)cosf(e);
      }
      pr[64 + 24 + c] = (_Float16)d3[c];
    }
  }
  __syncthreads();

  // ---- A-fragment base pointers (per k-step, per M-tile) ----
  const _Float16* aP0[10];
  const _Float16* aP1[10];
  const int r0 = m16, r1 = 16 + m16;

  auto setBuf = [&](const _Float16* buf) {
#pragma unroll
    for (int ks = 0; ks < 8; ++ks) {
      aP0[ks] = buf + (size_t)r0 * STR + ks * 32 + hi * 8;
      aP1[ks] = buf + (size_t)r1 * STR + ks * 32 + hi * 8;
    }
  };
  auto setPe = [&](int slot, int col) {
    aP0[slot] = sP + (size_t)r0 * STRP + col + hi * 8;
    aP1[slot] = sP + (size_t)r1 * STRP + col + hi * 8;
  };

  // ---- packed weight / bias pointers (tile offsets fixed by model shape) ----
  const _Float16* W1 = wts + (size_t)   0 * 512;  const float* B1 = bias +    0;
  const _Float16* W2 = wts + (size_t)  32 * 512;  const float* B2 = bias +  256;
  const _Float16* W3 = wts + (size_t) 160 * 512;  const float* B3 = bias +  512;
  const _Float16* W4 = wts + (size_t) 288 * 512;  const float* B4 = bias +  768;
  const _Float16* W5 = wts + (size_t) 416 * 512;  const float* B5 = bias + 1024;
  const _Float16* W6 = wts + (size_t) 544 * 512;  const float* B6 = bias + 1280;
  const _Float16* W7 = wts + (size_t) 704 * 512;  const float* B7 = bias + 1536;
  const _Float16* W8 = wts + (size_t) 832 * 512;  const float* B8 = bias + 1792;
  const _Float16* Wd = wts + (size_t) 960 * 512;  const float* Bd = bias + 2048;
  const _Float16* Wb = wts + (size_t) 968 * 512;  const float* Bb = bias + 2064;
  const _Float16* Wc1= wts + (size_t)1032 * 512;  const float* Bc1= bias + 2192;
  const _Float16* Wc = wts + (size_t)1072 * 512;  const float* Bc = bias + 2320;

  // ---- trunk ----
  setPe(0, 0); setPe(1, 32);                       // pe (K padded 39->64)
  gemmTile<2,  true>(W1, B1, 16, aP0, aP1, sA, lane); __syncthreads();
  setBuf(sA); gemmTile<8, true>(W2, B2, 16, aP0, aP1, sB, lane); __syncthreads();
  setBuf(sB); gemmTile<8, true>(W3, B3, 16, aP0, aP1, sA, lane); __syncthreads();
  setBuf(sA); gemmTile<8, true>(W4, B4, 16, aP0, aP1, sB, lane); __syncthreads();
  setBuf(sB); gemmTile<8, true>(W5, B5, 16, aP0, aP1, sA, lane); __syncthreads();
  setBuf(sA); setPe(8, 0); setPe(9, 32);           // skip: concat(x, pe), K 295->320
  gemmTile<10, true>(W6, B6, 16, aP0, aP1, sB, lane); __syncthreads();
  setBuf(sB); gemmTile<8, true>(W7, B7, 16, aP0, aP1, sA, lane); __syncthreads();
  setBuf(sA); gemmTile<8, true>(W8, B8, 16, aP0, aP1, sB, lane); __syncthreads();

  // ---- density head: 1 out channel (padded to 16), relu ----
  setBuf(sB);
  {
    v8f a0 = {}, a1 = {};
    const _Float16* wp = Wd + lane * 16;
#pragma unroll
    for (int ks = 0; ks < 8; ++ks) {
      v16h b = *(const v16h*)(wp + (size_t)ks * 512);
      a0 = wmma_f16(loadA(aP0[ks]), b, a0);
      a1 = wmma_f16(loadA(aP1[ks]), b, a1);
    }
    if (m16 == 0) {
      const float bv = Bd[0];
#pragma unroll
      for (int r = 0; r < 8; ++r) {
        int i0 = base + hi * 8 + r, i1 = base + 16 + hi * 8 + r;
        if (i0 < npts) out[i0] = fmaxf(a0[r] + bv, 0.f);
        if (i1 < npts) out[i1] = fmaxf(a1[r] + bv, 0.f);
      }
    }
  }

  // ---- bottleneck (128 outs) -> sA cols 0..127 ----
  gemmTile<8, true>(Wb, Bb, 8, aP0, aP1, sA, lane); __syncthreads();

  // ---- color layer 1: concat(bottleneck, de), K 155->160 ----
  setBuf(sA); setPe(4, 64);
  gemmTile<5, true>(Wc1, Bc1, 8, aP0, aP1, sB, lane); __syncthreads();

  // ---- color out: 3 channels (padded to 16), sigmoid ----
  setBuf(sB);
  {
    v8f a0 = {}, a1 = {};
    const _Float16* wp = Wc + lane * 16;
#pragma unroll
    for (int ks = 0; ks < 4; ++ks) {
      v16h b = *(const v16h*)(wp + (size_t)ks * 512);
      a0 = wmma_f16(loadA(aP0[ks]), b, a0);
      a1 = wmma_f16(loadA(aP1[ks]), b, a1);
    }
    if (m16 < 3) {
      const float bv = Bc[m16];
#pragma unroll
      for (int r = 0; r < 8; ++r) {
        int i0 = base + hi * 8 + r, i1 = base + 16 + hi * 8 + r;
        float v0 = 1.f / (1.f + expf(-(a0[r] + bv)));
        float v1 = 1.f / (1.f + expf(-(a1[r] + bv)));
        if (i0 < npts) out[(size_t)npts + (size_t)i0 * 3 + m16] = v0;
        if (i1 < npts) out[(size_t)npts + (size_t)i1 * 3 + m16] = v1;
      }
    }
  }
}

// ---------------------------------------------------------------------------
// Weight repack: f32 torch [N][K] -> f16 WMMA-B-fragment-ordered 16x32 tiles.
// Tile index = nt*kTiles + kt; within tile: lane*16 + e, element (lane,e) maps
// to k = kt*32 + (lane>>4)*8 + (e&7) + (e>>3)*16, n = nt*16 + (lane&15).  Pads with zeros.
// ---------------------------------------------------------------------------
__global__ void pack_layer(const float* __restrict__ W, const float* __restrict__ b,
                           _Float16* __restrict__ dw, float* __restrict__ db,
                           int Ksrc, int Nsrc, int kTiles, int nTiles)
{
  int gid = blockIdx.x * blockDim.x + threadIdx.x;
  int nThr = nTiles * kTiles * 32;
  if (gid < nThr) {
    int lane = gid & 31, tile = gid >> 5;
    int kt = tile % kTiles, nt = tile / kTiles;
    int n  = nt * 16 + (lane & 15);
    int kb = kt * 32 + (lane >> 4) * 8;
    _Float16* d = dw + (size_t)tile * 512 + lane * 16;
#pragma unroll
    for (int e = 0; e < 16; ++e) {
      int k = kb + (e & 7) + (e >> 3) * 16;
      float v = (k < Ksrc && n < Nsrc) ? W[(size_t)n * Ksrc + k] : 0.f;
      d[e] = (_Float16)v;
    }
  }
  int npad = nTiles * 16;
  if (gid < npad) db[gid] = (gid < Nsrc) ? b[gid] : 0.f;
}

// ---------------------------------------------------------------------------
extern "C" void kernel_launch(void* const* d_in, const int* in_sizes, int n_in,
                              void* d_out, int out_size, void* d_ws, size_t ws_size,
                              hipStream_t stream) {
  (void)n_in; (void)out_size; (void)ws_size;
  const float* pts  = (const float*)d_in[0];
  const float* dirs = (const float*)d_in[1];
  int npts = in_sizes[0] / 3;
  int nb   = in_sizes[1] / 3;
  int S    = (nb > 0) ? (npts / nb) : 1;

  _Float16* w  = (_Float16*)d_ws;
  float*    bb = (float*)((char*)d_ws + (size_t)1076 * 512 * 2);  // after 1076 1KB tiles

  struct L { int wi, bi, K, N, kt, nt, toff, boff; };
  const L Ls[12] = {
    { 2, 3,  39, 256,  2, 16,    0,    0},   // W1
    { 4, 5, 256, 256,  8, 16,   32,  256},   // W2
    { 6, 7, 256, 256,  8, 16,  160,  512},   // W3
    { 8, 9, 256, 256,  8, 16,  288,  768},   // W4
    {10,11, 256, 256,  8, 16,  416, 1024},   // W5
    {12,13, 295, 256, 10, 16,  544, 1280},   // W6 (skip)
    {14,15, 256, 256,  8, 16,  704, 1536},   // W7
    {16,17, 256, 256,  8, 16,  832, 1792},   // W8
    {18,19, 256,   1,  8,  1,  960, 2048},   // Wd
    {20,21, 256, 128,  8,  8,  968, 2064},   // Wb
    {22,23, 155, 128,  5,  8, 1032, 2192},   // Wc1
    {24,25, 128,   3,  4,  1, 1072, 2320},   // Wc
  };
  for (int i = 0; i < 12; ++i) {
    const L& l = Ls[i];
    int nThr = l.nt * l.kt * 32;
    int blocks = (nThr + 255) / 256;
    pack_layer<<<blocks, 256, 0, stream>>>(
        (const float*)d_in[l.wi], (const float*)d_in[l.bi],
        w + (size_t)l.toff * 512, bb + l.boff,
        l.K, l.N, l.kt, l.nt);
  }

  int blocks = (npts + 31) / 32;
  nerf_fused<<<blocks, 32, 0, stream>>>(pts, dirs, w, bb, (float*)d_out, npts, S);
}
